// DNDT_71519795413293
// MI455X (gfx1250) — compile-verified, matrix operand order checked by reading, not run on hardware
//
#include <hip/hip_runtime.h>
#include <hip/hip_fp16.h>

typedef __attribute__((ext_vector_type(16))) _Float16 v16h;
typedef __attribute__((ext_vector_type(8)))  float    v8f;
typedef __attribute__((ext_vector_type(2)))  _Float16 h2v;

#define TEMP_INV   10.0f      // 1 / 0.1
#define NFEAT      6
#define NCUT       3
#define NCLASS     10
#define NLEAF      4096
#define KCHUNKS    128        // 4096 / 32
#define WAVES_PB   8
#define ROWS_PB    128        // 8 waves * 16 rows
#define BSW_BYTES  131072     // 128 chunks * 32 lanes * 8 u32 pairs
#define HQ_BYTES   32768      // 8 waves * (16*64 h + 16*64 q) f16
#define LDS_BYTES  (BSW_BYTES + HQ_BYTES)

union FragU {
    v16h v;
    h2v  p[8];
    uint4 q[2];
};

extern "C" __global__ __launch_bounds__(256)
void dndt_fused(const float* __restrict__ x,
                const float* __restrict__ cuts,
                const float* __restrict__ S,
                float* __restrict__ out)
{
    extern __shared__ char smem[];
    unsigned int* Bsw = (unsigned int*)smem;             // [128][32][8] f16-pairs
    _Float16*     hq  = (_Float16*)(smem + BSW_BYTES);   // per wave: 1024 h + 1024 q f16

    const int tid    = threadIdx.x;
    const int wave   = tid >> 5;
    const int lane   = tid & 31;
    const int laneHi = lane >> 4;
    const int ncol   = lane & 15;

    // ---------- Stage 1: swizzle leaf_score (f32 [4096][10]) -> LDS f16 B fragments ----
    // Layout: Bsw[t][lane][v] = pair (S[k0, n], S[k0+1, n]) with n = lane&15 (0 pad n>=10)
    // k0 = t*32 + koff, koff = (v<4 ? 2v : 16+2(v-4)) + (lane>=16 ? 8 : 0)
    #pragma unroll 4
    for (int idx = tid; idx < KCHUNKS * 32 * 8; idx += 256) {
        const int t   = idx >> 8;
        const int ln  = (idx >> 3) & 31;
        const int v   = idx & 7;
        const int lhi = ln >> 4;
        const int nn  = ln & 15;
        const int koff = ((v < 4) ? (2 * v) : (16 + 2 * (v - 4))) + (lhi ? 8 : 0);
        const int k0   = t * 32 + koff;
        float f0 = 0.0f, f1 = 0.0f;
        if (nn < NCLASS) {
            f0 = S[k0 * NCLASS + nn];
            f1 = S[(k0 + 1) * NCLASS + nn];
        }
        union { h2v h; unsigned int u; } pk;
        pk.h[0] = (_Float16)f0;
        pk.h[1] = (_Float16)f1;
        Bsw[idx] = pk.u;
    }

    // ---------- Stage 2: soft binning + half-Kronecker into LDS ----------------------
    // lanes 0-15: h = p0 (x) p1 (x) p2 for row (lane&15); lanes 16-31: q = p3 (x) p4 (x) p5
    const int rowInWave = lane & 15;
    const int tileBase  = (blockIdx.x * WAVES_PB + wave) * 16;
    const int globalRow = tileBase + rowInWave;
    const int fbase     = laneHi * 3;

    float p[3][4];
    #pragma unroll
    for (int j = 0; j < 3; ++j) {
        const int f = fbase + j;
        float c0 = cuts[f * NCUT + 0];
        float c1 = cuts[f * NCUT + 1];
        float c2 = cuts[f * NCUT + 2];
        // sort3 ascending
        float a  = fminf(c0, c1), b = fmaxf(c0, c1);
        float s2 = fmaxf(b, c2);  float bm = fminf(b, c2);
        float s0 = fminf(a, bm);  float s1 = fmaxf(a, bm);
        const float b1 = -s0, b2 = -s0 - s1, b3 = -s0 - s1 - s2;
        const float xv = x[globalRow * NFEAT + f];
        float l0 = (xv * 1.0f     ) * TEMP_INV;
        float l1 = (xv * 2.0f + b1) * TEMP_INV;
        float l2 = (xv * 3.0f + b2) * TEMP_INV;
        float l3 = (xv * 4.0f + b3) * TEMP_INV;
        float m  = fmaxf(fmaxf(l0, l1), fmaxf(l2, l3));
        float e0 = __expf(l0 - m), e1 = __expf(l1 - m);
        float e2 = __expf(l2 - m), e3 = __expf(l3 - m);
        float inv = 1.0f / (e0 + e1 + e2 + e3);
        p[j][0] = e0 * inv; p[j][1] = e1 * inv;
        p[j][2] = e2 * inv; p[j][3] = e3 * inv;
    }
    // 64-wide kron of 3x 4-vectors, index = d0*16 + d1*4 + d2 (matches leaf>>6 / leaf&63)
    _Float16* dst = hq + wave * 2048 + laneHi * 1024 + rowInWave * 64;
    #pragma unroll
    for (int i = 0; i < 4; ++i) {
        #pragma unroll
        for (int j = 0; j < 4; ++j) {
            const float pij = p[0][i] * p[1][j];
            #pragma unroll
            for (int k = 0; k < 4; ++k)
                dst[i * 16 + j * 4 + k] = (_Float16)(pij * p[2][k]);
        }
    }

    __syncthreads();

    // ---------- Stage 3: K=4096 WMMA chain, 32 leaves per step ------------------------
    const _Float16* hrow = hq + wave * 2048 + rowInWave * 64;         // h[64]
    const _Float16* qrow = hrow + 1024;                               // q[64]

    v8f acc = {};
    #pragma unroll 2
    for (int t = 0; t < KCHUNKS; ++t) {
        // A fragment: leaf[row, t*32 + koff] = h[t>>1] * q[(t&1)*32 + koff]
        const _Float16 hv = hrow[t >> 1];
        const h2v hh = { hv, hv };
        const _Float16* qp = qrow + (t & 1) * 32 + laneHi * 8;
        FragU Q, A, B;
        Q.q[0] = *(const uint4*)(qp);          // pairs v=0..3 (K off 0..7  +laneHi*8)
        Q.q[1] = *(const uint4*)(qp + 16);     // pairs v=4..7 (K off 16..23+laneHi*8)
        #pragma unroll
        for (int v = 0; v < 8; ++v)
            A.p[v] = Q.p[v] * hh;              // v_pk_mul_f16

        const uint4* bp = (const uint4*)(Bsw + (t * 32 + lane) * 8);
        B.q[0] = bp[0];
        B.q[1] = bp[1];

        acc = __builtin_amdgcn_wmma_f32_16x16x32_f16(
            /*neg_a=*/false, A.v, /*neg_b=*/false, B.v,
            /*c_mod=*/(short)0, acc, /*reuse_a=*/false, /*reuse_b=*/false);
    }

    // ---------- Stage 4: store C (VGPR r -> M = r + 8*laneHi, N = lane&15) ------------
    if (ncol < NCLASS) {
        float* orow = out + (size_t)(tileBase + laneHi * 8) * NCLASS + ncol;
        #pragma unroll
        for (int r = 0; r < 8; ++r)
            orow[r * NCLASS] = acc[r];
    }
}

extern "C" void kernel_launch(void* const* d_in, const int* in_sizes, int n_in,
                              void* d_out, int out_size, void* d_ws, size_t ws_size,
                              hipStream_t stream) {
    const float* x    = (const float*)d_in[0];   // [32768, 6]
    const float* cuts = (const float*)d_in[1];   // [6, 3]
    const float* S    = (const float*)d_in[2];   // [4096, 10]
    float* out        = (float*)d_out;           // [32768, 10]

    (void)d_ws; (void)ws_size; (void)n_in; (void)out_size;

    // 160 KB dynamic LDS (> default cap) — raise the per-kernel limit every call
    // (deterministic, not a stream op, safe under graph capture).
    hipFuncSetAttribute((const void*)dndt_fused,
                        hipFuncAttributeMaxDynamicSharedMemorySize, LDS_BYTES);

    const int batch  = in_sizes[0] / NFEAT;      // 32768
    const int blocks = batch / ROWS_PB;          // 256
    dndt_fused<<<blocks, 256, LDS_BYTES, stream>>>(x, cuts, S, out);
}
